// NATBlock2D_84224308675133
// MI455X (gfx1250) — compile-verified, matrix operand order checked by reading, not run on hardware
//
#include <hip/hip_runtime.h>
#include <hip/hip_bf16.h>

typedef __attribute__((ext_vector_type(16))) __bf16 v16bf;
typedef __attribute__((ext_vector_type(8)))  __bf16 v8bf;
typedef __attribute__((ext_vector_type(8)))  float  v8f;

#define DEV static __device__ __forceinline__

DEV __bf16 f2bf(float f) {
  unsigned u = __builtin_bit_cast(unsigned, f);
  u += 0x7FFFu + ((u >> 16) & 1u);           // round-to-nearest-even
  unsigned short s = (unsigned short)(u >> 16);
  return __builtin_bit_cast(__bf16, s);
}

DEV v16bf cat16(v8bf lo, v8bf hi) {
  return __builtin_shufflevector(lo, hi, 0,1,2,3,4,5,6,7,8,9,10,11,12,13,14,15);
}

// ---------------------------------------------------------------------------
// fp32 -> bf16 elementwise convert (weights)
// ---------------------------------------------------------------------------
__global__ __launch_bounds__(256)
void nat_cvt_bf16(const float* __restrict__ s, __bf16* __restrict__ d, int n) {
  int i = blockIdx.x * 256 + threadIdx.x;
  if (i < n) d[i] = f2bf(s[i]);
}

// ---------------------------------------------------------------------------
// LayerNorm over C=256, one wave32 per row (8 values/lane), bf16 output
// ---------------------------------------------------------------------------
__global__ __launch_bounds__(256)
void nat_ln(const float* __restrict__ x, const float* __restrict__ g,
            const float* __restrict__ bta, __bf16* __restrict__ out, int rows) {
  const int row  = blockIdx.x * 8 + (threadIdx.x >> 5);
  const int lane = threadIdx.x & 31;
  if (row >= rows) return;
  const float* xr = x + (size_t)row * 256;
  float v[8];
#pragma unroll
  for (int t = 0; t < 8; ++t) v[t] = xr[lane + t * 32];
  float s = 0.f;
#pragma unroll
  for (int t = 0; t < 8; ++t) s += v[t];
#pragma unroll
  for (int off = 16; off; off >>= 1) s += __shfl_xor(s, off, 32);
  const float mean = s * (1.f / 256.f);
  float vs = 0.f;
#pragma unroll
  for (int t = 0; t < 8; ++t) { float d = v[t] - mean; vs += d * d; }
#pragma unroll
  for (int off = 16; off; off >>= 1) vs += __shfl_xor(vs, off, 32);
  const float rstd = rsqrtf(vs * (1.f / 256.f) + 1e-5f);
  __bf16* orow = out + (size_t)row * 256;
#pragma unroll
  for (int t = 0; t < 8; ++t) {
    const int cc = lane + t * 32;
    orow[cc] = f2bf((v[t] - mean) * rstd * g[cc] + bta[cc]);
  }
}

// ---------------------------------------------------------------------------
// bf16 WMMA GEMM: C[M,N] = A[M,K] @ W[N,K]^T (+bias, opt GELU, opt residual)
// block = 256 threads (8 waves), tile 128(M) x 64(N), K-step 32.
// Each wave owns a 32x32 output tile -> 2x2 of 16x16 WMMA accumulators.
// ---------------------------------------------------------------------------
#define BM 128
#define BN 64
#define BK 32
#define BKP 40   // padded LDS row stride (80B, multiple of 16B, spreads banks)

__global__ __launch_bounds__(256)
void nat_gemm(const __bf16* __restrict__ A, const __bf16* __restrict__ W,
              const float* __restrict__ bias, const float* __restrict__ resid,
              float* __restrict__ outF, __bf16* __restrict__ outBF,
              int M, int N, int K, int gelu) {
  __shared__ __align__(16) __bf16 As[BM][BKP];
  __shared__ __align__(16) __bf16 Bs[BN][BKP];

  const int tid  = threadIdx.x;
  const int m0   = blockIdx.y * BM;
  const int n0   = blockIdx.x * BN;
  const int wave = tid >> 5;
  const int lane = tid & 31;
  const int half = lane >> 4;     // which 16-lane half
  const int r    = lane & 15;     // row (M) for A-frags, col (N) for B-frags
  const int wm   = (wave & 3) * 32;
  const int wn   = (wave >> 2) * 32;

  v8f acc00 = {}, acc01 = {}, acc10 = {}, acc11 = {};

  // global->LDS assignments (even tiles, no edge cases: M%128==N%64==K%32==0)
  const int arow = tid >> 1;          // 0..127
  const int acol = (tid & 1) * 16;    // 0 / 16
  const int brow = tid >> 2;          // 0..63
  const int bcol = (tid & 3) * 8;     // 0/8/16/24

  for (int k0 = 0; k0 < K; k0 += BK) {
    const v8bf* ga = (const v8bf*)(A + (size_t)(m0 + arow) * K + k0 + acol);
    *(v8bf*)&As[arow][acol]     = ga[0];
    *(v8bf*)&As[arow][acol + 8] = ga[1];
    *(v8bf*)&Bs[brow][bcol] = *(const v8bf*)(W + (size_t)(n0 + brow) * K + k0 + bcol);
    if (k0 + BK < K) {
      __builtin_prefetch(A + (size_t)(m0 + arow) * K + k0 + BK + acol, 0, 1);
      __builtin_prefetch(W + (size_t)(n0 + brow) * K + k0 + BK + bcol, 0, 1);
    }
    __syncthreads();

    // Fragment gather per CDNA5 16-bit layout: element e -> K=(e>>3)*16+half*8+(e&7)
    // = two contiguous 8-bf16 (16B) chunks per lane.
    v16bf a0 = cat16(*(const v8bf*)&As[wm + r][half * 8],
                     *(const v8bf*)&As[wm + r][16 + half * 8]);
    v16bf a1 = cat16(*(const v8bf*)&As[wm + 16 + r][half * 8],
                     *(const v8bf*)&As[wm + 16 + r][16 + half * 8]);
    v16bf b0 = cat16(*(const v8bf*)&Bs[wn + r][half * 8],
                     *(const v8bf*)&Bs[wn + r][16 + half * 8]);
    v16bf b1 = cat16(*(const v8bf*)&Bs[wn + 16 + r][half * 8],
                     *(const v8bf*)&Bs[wn + 16 + r][16 + half * 8]);

    acc00 = __builtin_amdgcn_wmma_f32_16x16x32_bf16(false, a0, false, b0, (short)0, acc00, false, false);
    acc01 = __builtin_amdgcn_wmma_f32_16x16x32_bf16(false, a0, false, b1, (short)0, acc01, false, false);
    acc10 = __builtin_amdgcn_wmma_f32_16x16x32_bf16(false, a1, false, b0, (short)0, acc10, false, false);
    acc11 = __builtin_amdgcn_wmma_f32_16x16x32_bf16(false, a1, false, b1, (short)0, acc11, false, false);
    __syncthreads();
  }

  // Epilogue. C/D layout: element i -> M = i + 8*half, N = r (within 16x16 tile).
  v8f accs[2][2] = {{acc00, acc01}, {acc10, acc11}};
#pragma unroll
  for (int ti = 0; ti < 2; ++ti) {
#pragma unroll
    for (int tj = 0; tj < 2; ++tj) {
      const int col = n0 + wn + tj * 16 + r;
      const float bsv = bias[col];
#pragma unroll
      for (int i = 0; i < 8; ++i) {
        const int row = m0 + wm + ti * 16 + 8 * half + i;
        float v = accs[ti][tj][i] + bsv;
        if (gelu) v = 0.5f * v * (1.0f + erff(v * 0.70710678118654752f));
        if (resid) v += resid[(size_t)row * N + col];
        if (outF) outF[(size_t)row * N + col] = v;
        else      outBF[(size_t)row * N + col] = f2bf(v);
      }
    }
  }
}

// ---------------------------------------------------------------------------
// Neighborhood attention: block = 1 pixel, wave = 1 head, lane = channel (dh=32).
// qkv layout [P, 3*256] with columns (which, head, ch). Output bf16 [P,256].
// ---------------------------------------------------------------------------
__global__ __launch_bounds__(256)
void nat_attn(const float* __restrict__ qkv, const float* __restrict__ rpb,
              __bf16* __restrict__ out) {
  __shared__ float logits[8][52];
  const int p = blockIdx.x;
  const int b = p >> 12;
  const int i = (p >> 6) & 63;
  const int j = p & 63;
  const int h = threadIdx.x >> 5;
  const int c = threadIdx.x & 31;

  int si = i - 3; si = si < 0 ? 0 : (si > 57 ? 57 : si);
  int sj = j - 3; sj = sj < 0 ? 0 : (sj > 57 ? 57 : sj);

  const float scale = 0.17677669529663689f;   // 32^-0.5
  const float q = qkv[(size_t)p * 768 + h * 32 + c] * scale;
  const float* rpb_h = rpb + h * 169;

  for (int n = 0; n < 49; ++n) {
    const int ni = si + n / 7, nj = sj + n % 7;
    const size_t np = ((size_t)b << 12) + (ni << 6) + nj;
    float d = q * qkv[np * 768 + 256 + h * 32 + c];
#pragma unroll
    for (int off = 16; off; off >>= 1) d += __shfl_xor(d, off, 32);
    if (c == 0)
      logits[h][n] = d + rpb_h[(ni - i + 6) * 13 + (nj - j + 6)];
  }
  __syncthreads();

  float mx = -3.4e38f;
  for (int n = 0; n < 49; ++n) mx = fmaxf(mx, logits[h][n]);
  float den = 0.f;
  for (int n = 0; n < 49; ++n) den += __expf(logits[h][n] - mx);
  const float rden = 1.f / den;

  float o = 0.f;
  for (int n = 0; n < 49; ++n) {
    const int ni = si + n / 7, nj = sj + n % 7;
    const size_t np = ((size_t)b << 12) + (ni << 6) + nj;
    const float w = __expf(logits[h][n] - mx) * rden;
    o = fmaf(w, qkv[np * 768 + 512 + h * 32 + c], o);
  }
  out[(size_t)p * 256 + h * 32 + c] = f2bf(o);
}

// ---------------------------------------------------------------------------
extern "C" void kernel_launch(void* const* d_in, const int* in_sizes, int n_in,
                              void* d_out, int out_size, void* d_ws, size_t ws_size,
                              hipStream_t stream) {
  const float* x      = (const float*)d_in[0];
  const float* n1g    = (const float*)d_in[1];
  const float* n1b    = (const float*)d_in[2];
  const float* qkv_w  = (const float*)d_in[3];
  const float* qkv_b  = (const float*)d_in[4];
  const float* rpb    = (const float*)d_in[5];
  const float* proj_w = (const float*)d_in[6];
  const float* proj_b = (const float*)d_in[7];
  const float* n2g    = (const float*)d_in[8];
  const float* n2b    = (const float*)d_in[9];
  const float* fc1_w  = (const float*)d_in[10];
  const float* fc1_b  = (const float*)d_in[11];
  const float* fc2_w  = (const float*)d_in[12];
  const float* fc2_b  = (const float*)d_in[13];

  const int P = 2 * 64 * 64;   // 8192 rows
  const int C = 256;

  char* ws = (char*)d_ws;
  size_t off = 0;
  auto take = [&](size_t bytes) -> void* {
    void* p = ws + off;
    off = (off + bytes + 255) & ~(size_t)255;
    return p;
  };
  __bf16* xn_bf    = (__bf16*)take((size_t)P * C * 2);          // LN1 out
  __bf16* qkvw_bf  = (__bf16*)take((size_t)768 * 256 * 2);
  __bf16* projw_bf = (__bf16*)take((size_t)256 * 256 * 2);
  __bf16* fc1w_bf  = (__bf16*)take((size_t)1024 * 256 * 2);
  __bf16* fc2w_bf  = (__bf16*)take((size_t)256 * 1024 * 2);
  float*  qkv_f    = (float*)take((size_t)P * 768 * 4);
  __bf16* attn_bf  = (__bf16*)take((size_t)P * C * 2);
  float*  y_f      = (float*)take((size_t)P * C * 4);
  __bf16* hn_bf    = (__bf16*)take((size_t)P * C * 2);
  __bf16* h1_bf    = (__bf16*)take((size_t)P * 1024 * 2);
  (void)ws_size; (void)in_sizes; (void)n_in; (void)out_size;

  const dim3 blk(256);

  // weight conversions
  nat_cvt_bf16<<<(768 * 256 + 255) / 256, blk, 0, stream>>>(qkv_w, qkvw_bf, 768 * 256);
  nat_cvt_bf16<<<(256 * 256 + 255) / 256, blk, 0, stream>>>(proj_w, projw_bf, 256 * 256);
  nat_cvt_bf16<<<(1024 * 256 + 255) / 256, blk, 0, stream>>>(fc1_w, fc1w_bf, 1024 * 256);
  nat_cvt_bf16<<<(256 * 1024 + 255) / 256, blk, 0, stream>>>(fc2_w, fc2w_bf, 256 * 1024);

  // LN1 -> bf16
  nat_ln<<<P / 8, blk, 0, stream>>>(x, n1g, n1b, xn_bf, P);

  // QKV GEMM: [8192,256] x [768,256]^T -> fp32 [8192,768]
  nat_gemm<<<dim3(768 / BN, P / BM), blk, 0, stream>>>(
      xn_bf, qkvw_bf, qkv_b, nullptr, qkv_f, nullptr, P, 768, 256, 0);

  // neighborhood attention -> bf16 [8192,256]
  nat_attn<<<P, blk, 0, stream>>>(qkv_f, rpb, attn_bf);

  // proj GEMM + residual(x) -> y fp32
  nat_gemm<<<dim3(256 / BN, P / BM), blk, 0, stream>>>(
      attn_bf, projw_bf, proj_b, x, y_f, nullptr, P, 256, 256, 0);

  // LN2 -> bf16
  nat_ln<<<P / 8, blk, 0, stream>>>(y_f, n2g, n2b, hn_bf, P);

  // fc1 GEMM + GELU -> bf16 [8192,1024]
  nat_gemm<<<dim3(1024 / BN, P / BM), blk, 0, stream>>>(
      hn_bf, fc1w_bf, fc1_b, nullptr, nullptr, h1_bf, P, 1024, 256, 1);

  // fc2 GEMM + residual(y) -> d_out fp32
  nat_gemm<<<dim3(256 / BN, P / BM), blk, 0, stream>>>(
      h1_bf, fc2w_bf, fc2_b, y_f, (float*)d_out, nullptr, P, 256, 1024, 0);
}
